// VectorQuantizerEMA_17179869991
// MI455X (gfx1250) — matrix operand order repaired and checked by reading, not compiled, not run on hardware
//
#include <hip/hip_runtime.h>
#include <hip/hip_bf16.h>
#include <math.h>

#define NROWS 16384     // 32*512
#define DIM   64
#define NCODE 8192

typedef __attribute__((ext_vector_type(16))) _Float16 v16h;
typedef __attribute__((ext_vector_type(8)))  _Float16 v8h;
typedef __attribute__((ext_vector_type(4)))  _Float16 v4h;
typedef __attribute__((ext_vector_type(8)))  float    v8f;
typedef __attribute__((ext_vector_type(4)))  float    f4;
typedef __attribute__((ext_vector_type(2)))  float    f2;

union U16 { v16h v; v8h h2[2]; };

// ---- init per-call scratch (counts + loss accumulator) ----------------------
__global__ void k_init(int* __restrict__ counts, float* __restrict__ lossAcc) {
    int i = blockIdx.x * blockDim.x + threadIdx.x;
    if (i < NCODE) counts[i] = 0;
    if (i == 0) *lossAcc = 0.0f;
}

// ---- fp32 -> f16 conversion of the flattened inputs -------------------------
__global__ __launch_bounds__(256) void k_prep_inputs(const float* __restrict__ in,
                                                     _Float16* __restrict__ x16) {
    int i = blockIdx.x * blockDim.x + threadIdx.x;       // one float4 per thread
    f4 v = ((const f4*)in)[i];
    v4h h = { (_Float16)v.x, (_Float16)v.y, (_Float16)v.z, (_Float16)v.w };
    ((v4h*)x16)[i] = h;
}

// ---- codebook: fp32 -> f16 + per-code squared norm --------------------------
__global__ __launch_bounds__(256) void k_prep_codebook(const float* __restrict__ cb,
                                                       _Float16* __restrict__ e16,
                                                       float* __restrict__ enorm) {
    int c = blockIdx.x * blockDim.x + threadIdx.x;       // one code per thread
    if (c >= NCODE) return;
    const f4* src = (const f4*)(cb + (size_t)c * DIM);
    v4h* dst = (v4h*)(e16 + (size_t)c * DIM);
    float nrm = 0.0f;
    #pragma unroll
    for (int j = 0; j < DIM / 4; ++j) {
        f4 v = src[j];
        nrm += v.x * v.x + v.y * v.y + v.z * v.z + v.w * v.w;
        v4h h = { (_Float16)v.x, (_Float16)v.y, (_Float16)v.z, (_Float16)v.w };
        dst[j] = h;
    }
    enorm[c] = nrm;
}

// ---- WMMA distance GEMM + streaming argmin ---------------------------------
// wave handles 64 rows (4 row-tiles of 16); loops over 512 tiles of 16 codes.
// argmin(||e||^2 - 2 x.e) == argmin of full squared distance.
__global__ __launch_bounds__(256) void k_argmin(const _Float16* __restrict__ X16,
                                                const _Float16* __restrict__ E16,
                                                const float* __restrict__ enorm,
                                                int* __restrict__ bestIdx) {
    const int lane = threadIdx.x & 31;
    const int wave = threadIdx.x >> 5;                   // 0..7
    const int n    = lane & 15;                          // column within tile / row M for A
    const int hi   = lane >> 4;                          // half-wave selector
    const int row0 = (blockIdx.x * 8 + wave) * 64;

    // A fragments: contiguous k order per half-wave, same as B.
    //  elements 0..15 -> K = h*32 + hi*16 + e
    U16 a[4][2];
    #pragma unroll
    for (int t = 0; t < 4; ++t) {
        const _Float16* rp = X16 + (size_t)(row0 + t * 16 + n) * DIM;
        #pragma unroll
        for (int h = 0; h < 2; ++h) {
            a[t][h].v = *(const v16h*)(rp + h * 32 + hi * 16);
        }
    }

    float best[4][8];
    int   bidx[4][8];
    #pragma unroll
    for (int t = 0; t < 4; ++t)
        #pragma unroll
        for (int r = 0; r < 8; ++r) { best[t][r] = 3.4e38f; bidx[t][r] = 0x7fffffff; }

    for (int kt = 0; kt < NCODE / 16; ++kt) {
        // B fragment: contiguous k order, same as A; element e -> K(dim) = hi*16 + e,
        // column = lane&15. Row-major codebook makes this a contiguous 32B load.
        const _Float16* cp = E16 + (size_t)(kt * 16 + n) * DIM;
        U16 b0, b1;
        b0.v = *(const v16h*)(cp + hi * 16);             // dims  0..31
        b1.v = *(const v16h*)(cp + 32 + hi * 16);        // dims 32..63
        const float en = enorm[kt * 16 + n];
        const int   colIdx = kt * 16 + n;

        #pragma unroll
        for (int t = 0; t < 4; ++t) {
            v8f c = {0.f, 0.f, 0.f, 0.f, 0.f, 0.f, 0.f, 0.f};
            c = __builtin_amdgcn_wmma_f32_16x16x32_f16(false, a[t][0].v, false, b0.v,
                                                       (short)0, c, false, false);
            c = __builtin_amdgcn_wmma_f32_16x16x32_f16(false, a[t][1].v, false, b1.v,
                                                       (short)0, c, false, false);
            // C layout: element r -> row M = hi*8 + r, column N = lane&15
            #pragma unroll
            for (int r = 0; r < 8; ++r) {
                float d = __builtin_fmaf(-2.0f, c[r], en);
                if (d < best[t][r]) { best[t][r] = d; bidx[t][r] = colIdx; }
            }
        }
    }

    // reduce across the 16 lanes of each half-wave (first-min-index tie-break)
    #pragma unroll
    for (int off = 1; off < 16; off <<= 1) {
        #pragma unroll
        for (int t = 0; t < 4; ++t)
            #pragma unroll
            for (int r = 0; r < 8; ++r) {
                float ov = __shfl_xor(best[t][r], off, 32);
                int   oi = __shfl_xor(bidx[t][r], off, 32);
                if (ov < best[t][r] || (ov == best[t][r] && oi < bidx[t][r])) {
                    best[t][r] = ov; bidx[t][r] = oi;
                }
            }
    }
    if (n == 0) {
        #pragma unroll
        for (int t = 0; t < 4; ++t)
            #pragma unroll
            for (int r = 0; r < 8; ++r)
                bestIdx[row0 + t * 16 + hi * 8 + r] = bidx[t][r];
    }
}

// ---- zero the 512MB one-hot region ------------------------------------------
// enc base is only 8B-aligned within d_out, so: 1 f2 head, b128 nontemporal
// body (16B aligned), 1 f2 tail. This is the memory-bound phase that sets
// total runtime (~512MB / 23.3 TB/s), so stores must be maximally wide.
__global__ __launch_bounds__(256) void k_zero_enc(float* __restrict__ enc) {
    const long total = (long)NROWS * NCODE;              // 134217728 floats
    const long n4    = (total - 4) / 4;                  // aligned float4 count
    f4* body = (f4*)(enc + 2);
    long i      = (long)blockIdx.x * blockDim.x + threadIdx.x;
    long stride = (long)gridDim.x * blockDim.x;
    f4 z = {0.f, 0.f, 0.f, 0.f};
    for (long j = i; j < n4; j += stride) __builtin_nontemporal_store(z, body + j);
    if (i == 0) {
        f2 z2 = {0.f, 0.f};
        __builtin_nontemporal_store(z2, (f2*)enc);                   // head 2 floats
        __builtin_nontemporal_store(z2, (f2*)(enc + total - 2));     // tail 2 floats
    }
}

// ---- gather quantized rows, scatter one-hot, loss + histogram ---------------
__global__ __launch_bounds__(256) void k_finalize(const float* __restrict__ inputs,
                                                  const float* __restrict__ codebook,
                                                  const int* __restrict__ bestIdx,
                                                  float* __restrict__ out_q,
                                                  float* __restrict__ enc,
                                                  int* __restrict__ counts,
                                                  float* __restrict__ lossAcc) {
    const int d   = threadIdx.x & 63;
    const int row = blockIdx.x * 4 + (threadIdx.x >> 6);
    const int idx = bestIdx[row];
    const size_t o = (size_t)row * DIM + d;
    float q = codebook[(size_t)idx * DIM + d];
    out_q[o] = q;                                       // straight-through value == q
    float diff = q - inputs[o];
    float s = diff * diff;
    #pragma unroll
    for (int off = 16; off > 0; off >>= 1) s += __shfl_down(s, off, 32);
    if ((threadIdx.x & 31) == 0) atomicAdd(lossAcc, s);
    if (d == 0) {
        atomicAdd(counts + idx, 1);
        enc[(size_t)row * NCODE + idx] = 1.0f;
    }
}

// ---- scalars: commitment loss + perplexity ----------------------------------
__global__ __launch_bounds__(256) void k_stats(const int* __restrict__ counts,
                                               const float* __restrict__ lossAcc,
                                               float* __restrict__ out_loss,
                                               float* __restrict__ out_perp) {
    __shared__ float red[256];
    float s = 0.0f;
    for (int k = threadIdx.x; k < NCODE; k += 256) {
        float p = (float)counts[k] * (1.0f / (float)NROWS);
        s += p * logf(p + 1e-10f);
    }
    red[threadIdx.x] = s;
    __syncthreads();
    for (int off = 128; off > 0; off >>= 1) {
        if ((int)threadIdx.x < off) red[threadIdx.x] += red[threadIdx.x + off];
        __syncthreads();
    }
    if (threadIdx.x == 0) {
        *out_loss = 0.25f * (*lossAcc) * (1.0f / ((float)NROWS * (float)DIM));
        *out_perp = expf(-red[0]);
    }
}

extern "C" void kernel_launch(void* const* d_in, const int* in_sizes, int n_in,
                              void* d_out, int out_size, void* d_ws, size_t ws_size,
                              hipStream_t stream) {
    const float* inputs   = (const float*)d_in[0];      // [32,512,64] fp32
    const float* codebook = (const float*)d_in[1];      // [8192,64]  fp32
    float* out = (float*)d_out;

    // output layout: [loss | quantized_st (N*D) | perplexity | encodings (N*K)]
    float* out_loss = out;
    float* out_q    = out + 1;
    float* out_perp = out + 1 + (size_t)NROWS * DIM;
    float* out_enc  = out + 2 + (size_t)NROWS * DIM;

    // workspace layout
    char* ws = (char*)d_ws;
    _Float16* X16   = (_Float16*)(ws);                                  // 2 MB
    _Float16* E16   = (_Float16*)(ws + 2097152);                        // 1 MB
    float*    enorm = (float*)   (ws + 2097152 + 1048576);              // 32 KB
    int*      bidx  = (int*)     (ws + 2097152 + 1048576 + 32768);      // 64 KB
    int*      cnts  = (int*)     (ws + 2097152 + 1048576 + 32768 + 65536);
    float*    lacc  = (float*)   (ws + 2097152 + 1048576 + 32768 + 65536 + 32768);

    k_init<<<32, 256, 0, stream>>>(cnts, lacc);
    k_prep_inputs<<<(NROWS * DIM / 4) / 256, 256, 0, stream>>>(inputs, X16);
    k_prep_codebook<<<NCODE / 256, 256, 0, stream>>>(codebook, E16, enorm);
    k_argmin<<<NROWS / 512, 256, 0, stream>>>(X16, E16, enorm, bidx);
    k_zero_enc<<<16384, 256, 0, stream>>>(out_enc);
    k_finalize<<<NROWS / 4, 256, 0, stream>>>(inputs, codebook, bidx, out_q,
                                              out_enc, cnts, lacc);
    k_stats<<<1, 256, 0, stream>>>(cnts, lacc, out_loss, out_perp);
}
